// Reduce_52432960749778
// MI455X (gfx1250) — matrix operand order repaired and checked by reading, not compiled, not run on hardware
//
#include <hip/hip_runtime.h>
#include <cmath>

#define NH      128
#define BATCH   16
#define KSZ     4
#define STRIDE  2
#define MCH     384            // 3*NH output channels
#define KRAW    516            // 129*4 (im2col K), col 516 = bias column
#define KPAD    544            // 17 * 32
#define KSTEPS  17
#define TILE_N  64             // output positions per workgroup
#define SB_STRIDE 552          // padded LDS row stride (ushorts), multiple of 8
#define SH_STRIDE 136          // staged input row stride (>=132), multiple of 8
#define SLICE_U (MCH * 32)     // ushorts per weight K-slice (12288 = 24KB)

typedef __attribute__((ext_vector_type(16))) __bf16 v16bf;
typedef __attribute__((ext_vector_type(8)))  float  v8f;

union BFrag { v16bf v; uint4 q[2]; };

__device__ __forceinline__ unsigned short f2bf(float f) {
  unsigned int x = __float_as_uint(f);
  x += 0x7FFFu + ((x >> 16) & 1u);       // round-to-nearest-even
  return (unsigned short)(x >> 16);
}

__device__ __forceinline__ int nstep(int n) {
  int m = n - KSZ; if (m < 0) m = 0;
  return (m + STRIDE - 1) / STRIDE + 1;
}

// LDS byte offset of a pointer into __shared__ memory (low 32 bits of generic)
__device__ __forceinline__ unsigned lds_off(const void* p) {
  return (unsigned)(uintptr_t)p;
}

// CDNA5 async global->LDS copy (ASYNCcnt-tracked), 16 bytes per lane issue
__device__ __forceinline__ void async_copy_b128(unsigned lds_byte_off,
                                                const void* gptr) {
  unsigned long long ga = (unsigned long long)(uintptr_t)gptr;
  asm volatile("global_load_async_to_lds_b128 %0, %1, off"
               :: "v"(lds_byte_off), "v"(ga) : "memory");
}

__device__ __forceinline__ void wait_asynccnt0() {
  asm volatile("s_wait_asynccnt 0x0" ::: "memory");
}

// ---------------------------------------------------------------- weights ---
// Output layout is K-slice-major: Wb[ks][row][col], each slice 384x32 bf16
// contiguous (24KB) so the per-step async stage is fully coalesced.
__global__ void prep_weights(const float* __restrict__ W,
                             const float* __restrict__ bias,
                             unsigned short* __restrict__ Wb) {
  int idx = blockIdx.x * blockDim.x + threadIdx.x;
  if (idx >= MCH * KPAD) return;
  int oc = idx / KPAD, kk = idx - oc * KPAD;
  float v = 0.f;
  if (kk < KRAW)       v = W[oc * KRAW + kk];   // W[oc][ic][k], kk = ic*4+k
  else if (kk == KRAW) v = bias[oc];            // bias column (B row = 1.0)
  int ks = kk >> 5, c = kk & 31;
  Wb[(size_t)ks * SLICE_U + oc * 32 + c] = f2bf(v);
}

// -------------------------------------------------------------- conv step ---
__global__ __launch_bounds__(256)
void conv_step(const float* __restrict__ hin, float* __restrict__ hout,
               const int* __restrict__ N0, const unsigned short* __restrict__ Wb,
               float* __restrict__ stage,
               int Lin, int Lout, int depth, float logd) {
  // sH (input staging) is dead after im2col; reuse its space for the
  // double-buffered async weight slices.
  __shared__ __align__(16) union {
    unsigned short sH[129 * SH_STRIDE];          // 35 KB
    unsigned short wS[2][SLICE_U];               // 2 x 24 KB
  } uS;
  __shared__ __align__(16) unsigned short sB[TILE_N * SB_STRIDE];  // 69 KB

  const int tid  = threadIdx.x;
  const int lane = tid & 31;
  const int wave = tid >> 5;
  const int b    = blockIdx.y;
  const int t0   = blockIdx.x * TILE_N;

  // N recurrence (per batch, closed form from N0)
  int ncur = N0[b];
  for (int d = 0; d < depth; ++d) ncur = nstep(ncur);
  const int  nnext = nstep(ncur);
  const bool newly = (nnext <= 1) && (ncur > 1 || depth == 0);

  // valid local input count: pos < min(Lin, N)  (mask + right-pad combined)
  const int limit = (Lin < ncur ? Lin : ncur) - 2 * t0;

  // phase 1: coalesced global -> LDS staging (bf16, masked), rows = channels
  for (int i = 0; i < 17; ++i) {
    int ic = wave + 8 * i;
    if (ic > 128) break;
    const float* row = hin + ((size_t)b * NH + ic) * (size_t)Lin + 2 * t0;
    for (int off = lane; off < SH_STRIDE; off += 32) {
      float f = 0.f;
      if (off < limit) f = (ic < NH) ? row[off] : logd;  // ic==128: depth chan
      uS.sH[ic * SH_STRIDE + off] = f2bf(f);
    }
  }
  __syncthreads();

  // phase 2: im2col  sH -> sB[n][kk]
  for (int idx = tid; idx < TILE_N * KPAD; idx += 256) {
    int n  = idx / KPAD;
    int kk = idx - n * KPAD;
    unsigned short u = 0;
    if (kk < KRAW) {
      int ic = kk >> 2, k = kk & 3;
      u = uS.sH[ic * SH_STRIDE + 2 * n + k];
    } else if (kk == KRAW) {
      u = 0x3F80;                       // 1.0 bf16 : bias row
    }
    sB[n * SB_STRIDE + kk] = u;
  }
  __syncthreads();                      // sH dead from here; wS takes over

  // wave tiling: nt = N-tile (4 x 16 cols), p = parity of M-tiles
  const int p  = wave >> 2;
  const int nt = wave & 3;

  v8f acc[12] = {};                     // j<4: l, 4..7: r, 8..11: g

  const int col   = 16 * nt + (lane & 15);
  const int halfB = (lane < 16) ? 0 : 16;   // B: K 0..15 / 16..31 contiguous
  const int halfA = (lane < 16) ? 0 : 16;   // A: K 0..15 / 16..31 contiguous
  const int arow  = lane & 15;

  // async stage of one 24KB weight K-slice: 1536 b128 chunks, 6 per thread
  auto stage_slice = [&](int buf, int ks) {
    const unsigned short* gsl = Wb + (size_t)ks * SLICE_U;
    unsigned lbase = lds_off(&uS.wS[buf][0]);
#pragma unroll
    for (int i = 0; i < 6; ++i) {
      int e = (tid + 256 * i) * 8;      // ushort offset of 16B chunk
      async_copy_b128(lbase + e * 2, gsl + e);
    }
  };

  stage_slice(0, 0);                    // prime the pipeline

  for (int ks = 0; ks < KSTEPS; ++ks) {
    const int cur = ks & 1;
    wait_asynccnt0();                   // slice ks landed (issued last iter)
    __syncthreads();                    // visible to all waves; prior readers done
    if (ks + 1 < KSTEPS) stage_slice(cur ^ 1, ks + 1);  // overlaps compute

    BFrag bf;
    const unsigned short* bp = sB + col * SB_STRIDE + 32 * ks + halfB;
    bf.q[0] = *(const uint4*)(bp);
    bf.q[1] = *(const uint4*)(bp + 8);

    const unsigned short* ws = &uS.wS[cur][0];
#pragma unroll
    for (int j = 0; j < 12; ++j) {
      const int mt = p + 2 * j;         // M-tile (parity-interleaved)
      BFrag af;
      const unsigned short* ap = ws + (16 * mt + arow) * 32 + halfA;
      af.q[0] = *(const uint4*)(ap);
      af.q[1] = *(const uint4*)(ap + 8);
      acc[j] = __builtin_amdgcn_wmma_f32_16x16x32_bf16(
          false, af.v, false, bf.v, (short)0, acc[j], false, false);
    }
  }

  // epilogue: gated combine, store h_new, capture finished batches at t==0
  const int hi = lane >> 4;
  const int t  = t0 + 16 * nt + (lane & 15);
#pragma unroll
  for (int j = 0; j < 4; ++j) {
    const int ct = p + 2 * j;           // channel tile 0..7
#pragma unroll
    for (int v = 0; v < 8; ++v) {
      const int c = 16 * ct + v + 8 * hi;
      const float l = acc[j][v];
      const float r = tanhf(acc[j + 4][v]);
      const float g = 1.f / (1.f + __expf(-acc[j + 8][v]));
      const float val = l * g + r * (1.f - g);
      if (t < Lout) hout[((size_t)b * NH + c) * (size_t)Lout + t] = val;
      if (newly && t == 0) stage[b * NH + c] = val;
    }
  }
}

// ---------------------------------------------------- finish-depth argsort ---
__global__ void gather_out(const float* __restrict__ stage,
                           const int* __restrict__ N0,
                           float* __restrict__ out) {
  int tid = blockIdx.x * blockDim.x + threadIdx.x;
  if (tid >= BATCH * NH) return;
  int b = tid / NH, c = tid - b * NH;
  int fin[BATCH];
  for (int j = 0; j < BATCH; ++j) {
    int n = N0[j], f = 12; bool done = false;
    for (int d = 0; d < 13; ++d) {
      int nc = n; n = nstep(n);
      if (!done && n <= 1 && (nc > 1 || d == 0)) { f = d; done = true; }
    }
    fin[j] = f;
  }
  int fb = fin[b], rank = 0;
  for (int j = 0; j < BATCH; ++j)
    rank += (fin[j] < fb) || (fin[j] == fb && j < b);
  out[rank * NH + c] = stage[b * NH + c];
}

// ------------------------------------------------------------------ launch ---
extern "C" void kernel_launch(void* const* d_in, const int* in_sizes, int n_in,
                              void* d_out, int out_size, void* d_ws, size_t ws_size,
                              hipStream_t stream) {
  const float* h  = (const float*)d_in[0];
  const int*   N0 = (const int*)d_in[1];
  const float* W  = (const float*)d_in[2];
  const float* bv = (const float*)d_in[3];

  // workspace layout: ping (even depths, <=8191) | pong (odd, <=4095) | Wb | stage
  float* buf0 = (float*)d_ws;
  float* buf1 = buf0 + (size_t)BATCH * NH * 8191;
  unsigned short* Wb = (unsigned short*)(buf1 + (size_t)BATCH * NH * 4095);
  float* stage = (float*)(Wb + (size_t)KSTEPS * SLICE_U);

  prep_weights<<<(MCH * KPAD + 255) / 256, 256, 0, stream>>>(W, bv, Wb);

  int L = 16384;
  const float* cur = h;
  float* bufs[2] = {buf0, buf1};
  for (int d = 0; d < 13; ++d) {
    int Lp = L;
    if (Lp < KSZ) Lp = KSZ;
    else if ((Lp - KSZ) % STRIDE) Lp += STRIDE - (Lp - KSZ) % STRIDE;
    const int Lout = (Lp - KSZ) / STRIDE + 1;
    float* outp = bufs[d & 1];
    dim3 grid((Lout + TILE_N - 1) / TILE_N, BATCH);
    conv_step<<<grid, 256, 0, stream>>>(cur, outp, N0, Wb, stage,
                                        L, Lout, d, log1pf((float)d));
    cur = outp;
    L = Lout;
  }
  gather_out<<<(BATCH * NH + 255) / 256, 256, 0, stream>>>(stage, N0, (float*)d_out);
}
